// SelfAttention_54546084659542
// MI455X (gfx1250) — matrix operand order repaired and checked by reading, not compiled, not run on hardware
//
#include <hip/hip_runtime.h>
#include <hip/hip_bf16.h>

// ---------------------------------------------------------------------------
// Self-attention (B=4, S=2048, C=256, H=8, head_dim=256) for gfx1250 (MI455X)
// Flash-attention formulation, all GEMMs via v_wmma_f32_16x16x32_f16.
// K/V tiles staged with GLOBAL_LOAD_ASYNC_TO_LDS_B128 (ASYNCcnt), double
// buffered so the DMA for tile kt+1 overlaps the WMMA work of tile kt.
// Workspace layout (f16): WqT,WkT,WvT,WoutT (4 x 512K halves) then
// Q,K,Vt,U (4 x 16.78M halves) = ~132 MB total.
// ---------------------------------------------------------------------------

typedef __attribute__((ext_vector_type(16))) _Float16 v16h;
typedef __attribute__((ext_vector_type(8)))  _Float16 v8h;
typedef __attribute__((ext_vector_type(8)))  float    v8f;

#define NB   4      // batch
#define SEQ  2048   // sequence length
#define CIN  256    // in channels == head dim
#define NH   8      // heads
#define HC   2048   // NH * CIN
#define SOFTMAX_SCALE 0.17677669529663687f   // 1/sqrt(256/8)

// D = A(16xK) * B(Kx16) + C, f16 inputs, f32 accumulate
static __device__ inline v8f wmma_f16(v16h a, v16h b, v8f c) {
  return __builtin_amdgcn_wmma_f32_16x16x32_f16(false, a, false, b,
                                                (short)0, c, false, false);
}

// A-fragment: lane m holds K = 8*half + [0..7] and 16 + 8*half + [0..7].
static __device__ inline v16h ld_afrag(const _Float16* p) {
  union { v16h v; v8h h[2]; } u;
  u.h[0] = *(const v8h*)(p);
  u.h[1] = *(const v8h*)(p + 16);
  return u.v;
}

// B-fragment: lane n holds K = 8*half + [0..7] and 16 + 8*half + [0..7].
static __device__ inline v16h ld_bfrag(const _Float16* p) {
  union { v16h v; v8h h[2]; } u;
  u.h[0] = *(const v8h*)(p);
  u.h[1] = *(const v8h*)(p + 16);
  return u.v;
}

// A-fragment built from f32 source (converted to f16 on the fly).
static __device__ inline v16h cvt_afrag(const float* p) {
  v16h a;
#pragma unroll
  for (int i = 0; i < 8; ++i) a[i] = (_Float16)p[i];
#pragma unroll
  for (int i = 0; i < 8; ++i) a[8 + i] = (_Float16)p[16 + i];
  return a;
}

// --- CDNA5 async direct-to-LDS copy (tracked by ASYNCcnt) ------------------
// Low 32 bits of a flat shared pointer == wave-relative LDS byte address.
static __device__ inline unsigned lds_addr32(const void* p) {
  return (unsigned)(uintptr_t)p;
}
static __device__ inline void async_ld_b128(unsigned lds_dst, const void* gsrc) {
  asm volatile("global_load_async_to_lds_b128 %0, %1, off"
               :: "v"(lds_dst), "v"((unsigned long long)(uintptr_t)gsrc)
               : "memory");
}
static __device__ inline void wait_async0() {
#if __has_builtin(__builtin_amdgcn_s_wait_asynccnt)
  __builtin_amdgcn_s_wait_asynccnt(0);
#else
  asm volatile("s_wait_asynccnt 0" ::: "memory");
#endif
}

// ---------------------------------------------------------------------------
// Kernel 0: dst[c*rows + r] = src[r*cols + c] * scale   (f32 -> f16 transpose)
// ---------------------------------------------------------------------------
__global__ void k_transpose_w(const float* __restrict__ src,
                              _Float16* __restrict__ dst,
                              int rows, int cols, float scale) {
  int idx = blockIdx.x * blockDim.x + threadIdx.x;
  if (idx >= rows * cols) return;
  int r = idx / cols, c = idx % cols;
  dst[(size_t)c * rows + r] = (_Float16)(src[idx] * scale);
}

// ---------------------------------------------------------------------------
// Kernel 1: QKV projections.  [8192 x 256] f32  x  WT[2048 x 256] f16.
// Q,K stored [b,h,s,c] f16;  V stored transposed [b,h,c,s] f16.
// ---------------------------------------------------------------------------
__global__ __launch_bounds__(128) void k_proj(
    const float* __restrict__ v,
    const _Float16* __restrict__ wqT, const _Float16* __restrict__ wkT,
    const _Float16* __restrict__ wvT,
    _Float16* __restrict__ Qh, _Float16* __restrict__ Kh,
    _Float16* __restrict__ Vth) {
  const int lane = threadIdx.x & 31, wave = threadIdx.x >> 5;
  const int n = lane & 15, half = lane >> 4;
  const int rowbase = blockIdx.x * 64 + wave * 16;

  v16h af[8];
  const float* vrow = v + (size_t)(rowbase + n) * CIN;
#pragma unroll
  for (int kc = 0; kc < 8; ++kc)
    af[kc] = cvt_afrag(vrow + kc * 32 + 8 * half);

  const _Float16* Wts[2] = {wqT, wkT};
  _Float16* Outs[2]      = {Qh, Kh};

  for (int w = 0; w < 2; ++w) {
    const _Float16* Wt = Wts[w];
    _Float16* Out = Outs[w];
    for (int nc = 0; nc < 128; ++nc) {
      v8f acc = {};
#pragma unroll
      for (int kc = 0; kc < 8; ++kc) {
        v16h bf = ld_bfrag(Wt + (size_t)(nc * 16 + n) * CIN + kc * 32 + 8 * half);
        acc = wmma_f16(af[kc], bf, acc);
      }
      const int h = nc >> 4;
      const int c = ((nc & 15) << 4) + n;
#pragma unroll
      for (int r = 0; r < 8; ++r) {
        int g = rowbase + r + 8 * half;
        int b = g >> 11, s = g & (SEQ - 1);
        Out[(((size_t)(b * NH + h) * SEQ + s) * CIN) + c] = (_Float16)acc[r];
      }
    }
  }

  for (int nc = 0; nc < 128; ++nc) {
    v8f acc = {};
#pragma unroll
    for (int kc = 0; kc < 8; ++kc) {
      v16h bf = ld_bfrag(wvT + (size_t)(nc * 16 + n) * CIN + kc * 32 + 8 * half);
      acc = wmma_f16(af[kc], bf, acc);
    }
    const int h = nc >> 4;
    const int c = ((nc & 15) << 4) + n;
#pragma unroll
    for (int r = 0; r < 8; ++r) {
      int g = rowbase + r + 8 * half;
      int b = g >> 11, s = g & (SEQ - 1);
      Vth[(((size_t)(b * NH + h) * CIN + c) * SEQ) + s] = (_Float16)acc[r];
    }
  }
}

// ---------------------------------------------------------------------------
// Kernel 2: flash attention per (b,h).  grid=(S/64, B*H), 4 waves x 16 rows.
// Double-buffered K [32 x 256] and Vt [256 x 32] tiles, async DMA to LDS.
// ---------------------------------------------------------------------------
#define LDK 264   // 256 + 8 halves padding (16B-aligned rows, bank-spread)
#define LDV 40    // 32 + 8 halves padding

static __device__ inline void stage_tiles_async(
    const _Float16* __restrict__ Kbase, const _Float16* __restrict__ Vbase,
    _Float16* KsBuf, _Float16* VsBuf, int kt, int tid) {
#pragma unroll
  for (int j = 0; j < 8; ++j) {
    int vec = tid + 128 * j;                 // 1024 vectors of 8 halves
    int r = vec >> 5, c = (vec & 31) * 8;
    async_ld_b128(lds_addr32(&KsBuf[r * LDK + c]),
                  Kbase + (size_t)(kt * 32 + r) * CIN + c);
  }
#pragma unroll
  for (int j = 0; j < 8; ++j) {
    int vec = tid + 128 * j;
    int r = vec >> 2, c = (vec & 3) * 8;
    async_ld_b128(lds_addr32(&VsBuf[r * LDV + c]),
                  Vbase + (size_t)r * SEQ + kt * 32 + c);
  }
}

__global__ __launch_bounds__(128) void k_attn(
    const _Float16* __restrict__ Qh, const _Float16* __restrict__ Kh,
    const _Float16* __restrict__ Vth, _Float16* __restrict__ Uh) {
  __shared__ _Float16 Ks[2][32 * LDK];
  __shared__ _Float16 Vs[2][256 * LDV];
  __shared__ _Float16 Ps[4][16 * LDV];

  const int tid = threadIdx.x, lane = tid & 31, wave = tid >> 5;
  const int n = lane & 15, half = lane >> 4;
  const int bh = blockIdx.y;
  const int qbase = blockIdx.x * 64 + wave * 16;

  v16h qf[8];
  const _Float16* qrow = Qh + ((size_t)bh * SEQ + qbase + n) * CIN;
#pragma unroll
  for (int kc = 0; kc < 8; ++kc)
    qf[kc] = ld_afrag(qrow + kc * 32 + 8 * half);

  v8f o[16];
#pragma unroll
  for (int i = 0; i < 16; ++i) o[i] = (v8f){};
  float mrow[8], lrow[8];
#pragma unroll
  for (int r = 0; r < 8; ++r) { mrow[r] = -INFINITY; lrow[r] = 0.0f; }

  const _Float16* Kbase = Kh  + (size_t)bh * SEQ * CIN;
  const _Float16* Vbase = Vth + (size_t)bh * CIN * SEQ;

  const int NT = SEQ / 32;
  // Prologue: DMA tile 0 into buffer 0.
  stage_tiles_async(Kbase, Vbase, &Ks[0][0], &Vs[0][0], 0, tid);

  for (int kt = 0; kt < NT; ++kt) {
    const int cur = kt & 1;
    // Tile kt's async ops (issued last iteration / prologue) must land.
    wait_async0();
    __syncthreads();
    // Overlap: DMA tile kt+1 into the other buffer while we compute on kt.
    if (kt + 1 < NT)
      stage_tiles_async(Kbase, Vbase, &Ks[cur ^ 1][0], &Vs[cur ^ 1][0],
                        kt + 1, tid);

    // ---- scores: two 16x16 accumulators over the 32-key tile ----
    v8f s0 = {}, s1 = {};
#pragma unroll
    for (int kc = 0; kc < 8; ++kc) {
      v16h b0 = ld_bfrag(&Ks[cur][(size_t)n * LDK + kc * 32 + 8 * half]);
      v16h b1 = ld_bfrag(&Ks[cur][(size_t)(16 + n) * LDK + kc * 32 + 8 * half]);
      s0 = wmma_f16(qf[kc], b0, s0);
      s1 = wmma_f16(qf[kc], b1, s1);
    }

    // ---- online softmax (row reductions across 16-lane groups) ----
    float pscale[8];
#pragma unroll
    for (int r = 0; r < 8; ++r) {
      float t = fmaxf(s0[r], s1[r]);
      t = fmaxf(t, __shfl_xor(t, 1));
      t = fmaxf(t, __shfl_xor(t, 2));
      t = fmaxf(t, __shfl_xor(t, 4));
      t = fmaxf(t, __shfl_xor(t, 8));
      float mnew = fmaxf(mrow[r], t);
      float sc = __expf(mrow[r] - mnew);
      float p0 = __expf(s0[r] - mnew);
      float p1 = __expf(s1[r] - mnew);
      float rs = p0 + p1;
      rs += __shfl_xor(rs, 1);
      rs += __shfl_xor(rs, 2);
      rs += __shfl_xor(rs, 4);
      rs += __shfl_xor(rs, 8);
      lrow[r] = lrow[r] * sc + rs;
      mrow[r] = mnew;
      pscale[r] = sc;
      // stage P row-major [16 rows][32 keys] for A-layout reload
      Ps[wave][(r + 8 * half) * LDV + n]      = (_Float16)p0;
      Ps[wave][(r + 8 * half) * LDV + 16 + n] = (_Float16)p1;
    }
    asm volatile("s_wait_dscnt 0" ::: "memory");   // wave-internal LDS RAW

    union { v16h v; v8h h[2]; } pu;
    pu.h[0] = *(const v8h*)&Ps[wave][n * LDV + 8 * half];
    pu.h[1] = *(const v8h*)&Ps[wave][n * LDV + 16 + 8 * half];
    v16h pf = pu.v;

    // ---- rescale + accumulate PV over all 256 channels ----
#pragma unroll
    for (int nc = 0; nc < 16; ++nc) {
      v8f acc = o[nc];
#pragma unroll
      for (int r = 0; r < 8; ++r) acc[r] *= pscale[r];
      v16h bv = ld_bfrag(&Vs[cur][(size_t)(nc * 16 + n) * LDV + 8 * half]);
      o[nc] = wmma_f16(pf, bv, acc);
    }
    // No trailing barrier: next iteration's barrier (after wait_async0)
    // collectively orders these reads before any write to this buffer,
    // which can only happen at tile kt+2's stage call.
  }

  // ---- epilogue: normalize and store U [B*S][H*C] as f16 ----
  const int b = bh / NH, h = bh % NH;
  float inv[8];
#pragma unroll
  for (int r = 0; r < 8; ++r) inv[r] = 1.0f / lrow[r];
  const size_t rowg0 = (size_t)b * SEQ + qbase;
#pragma unroll
  for (int nc = 0; nc < 16; ++nc) {
    int c = h * CIN + nc * 16 + n;
#pragma unroll
    for (int r = 0; r < 8; ++r)
      Uh[(rowg0 + r + 8 * half) * HC + c] = (_Float16)(o[nc][r] * inv[r]);
  }
}

// ---------------------------------------------------------------------------
// Kernel 3: out = U[8192 x 2048] x Wout[2048 x 256] + bout, f32 result.
// ---------------------------------------------------------------------------
__global__ __launch_bounds__(128) void k_outproj(
    const _Float16* __restrict__ Uh, const _Float16* __restrict__ woT,
    const float* __restrict__ bout, float* __restrict__ out) {
  const int lane = threadIdx.x & 31, wave = threadIdx.x >> 5;
  const int n = lane & 15, half = lane >> 4;
  const int rowbase = blockIdx.x * 64 + wave * 16;

  v8f acc[16];
#pragma unroll
  for (int i = 0; i < 16; ++i) acc[i] = (v8f){};

  const _Float16* urow = Uh + (size_t)(rowbase + n) * HC;
  for (int kc = 0; kc < HC / 32; ++kc) {
    v16h af = ld_afrag(urow + kc * 32 + 8 * half);
#pragma unroll
    for (int nc = 0; nc < 16; ++nc) {
      v16h bf = ld_bfrag(woT + (size_t)(nc * 16 + n) * HC + kc * 32 + 8 * half);
      acc[nc] = wmma_f16(af, bf, acc[nc]);
    }
  }
#pragma unroll
  for (int nc = 0; nc < 16; ++nc) {
    float bias = bout[nc * 16 + n];
#pragma unroll
    for (int r = 0; r < 8; ++r)
      out[(size_t)(rowbase + r + 8 * half) * CIN + nc * 16 + n] =
          acc[nc][r] + bias;
  }
}

// ---------------------------------------------------------------------------
extern "C" void kernel_launch(void* const* d_in, const int* in_sizes, int n_in,
                              void* d_out, int out_size, void* d_ws,
                              size_t ws_size, hipStream_t stream) {
  const float* v    = (const float*)d_in[0];
  const float* WQ   = (const float*)d_in[1];
  const float* WK   = (const float*)d_in[2];
  const float* WV   = (const float*)d_in[3];
  const float* Wout = (const float*)d_in[4];
  const float* bout = (const float*)d_in[5];

  _Float16* ws = (_Float16*)d_ws;
  const size_t WSZ = (size_t)CIN * HC;             // 524288 halves per weight
  const size_t TSZ = (size_t)NB * NH * SEQ * CIN;  // 16,777,216 halves
  _Float16* wqT = ws;
  _Float16* wkT = wqT + WSZ;
  _Float16* wvT = wkT + WSZ;
  _Float16* woT = wvT + WSZ;
  _Float16* Qh  = woT + WSZ;
  _Float16* Kh  = Qh + TSZ;
  _Float16* Vth = Kh + TSZ;
  _Float16* Uh  = Vth + TSZ;   // total ~132 MB of workspace used

  int wthreads = 256, wblocks = (int)(WSZ / wthreads);
  k_transpose_w<<<wblocks, wthreads, 0, stream>>>(WQ, wqT, CIN, HC, SOFTMAX_SCALE);
  k_transpose_w<<<wblocks, wthreads, 0, stream>>>(WK, wkT, CIN, HC, 1.0f);
  k_transpose_w<<<wblocks, wthreads, 0, stream>>>(WV, wvT, CIN, HC, 1.0f);
  k_transpose_w<<<wblocks, wthreads, 0, stream>>>(Wout, woT, HC, CIN, 1.0f);

  k_proj<<<dim3((NB * SEQ) / 64), dim3(128), 0, stream>>>(v, wqT, wkT, wvT,
                                                          Qh, Kh, Vth);
  k_attn<<<dim3(SEQ / 64, NB * NH), dim3(128), 0, stream>>>(Qh, Kh, Vth, Uh);
  k_outproj<<<dim3((NB * SEQ) / 64), dim3(128), 0, stream>>>(Uh, woT, bout,
                                                             (float*)d_out);
}
